// T2SBlock_30614526886317
// MI455X (gfx1250) — matrix operand order repaired and checked by reading, not compiled, not run on hardware
//
#include <hip/hip_runtime.h>
#include <hip/hip_bf16.h>

// ---------------------------------------------------------------------------
// MI455X (gfx1250, wave32) transformer block.
// All GEMMs run bf16 WMMA (f32 accum) on pre-converted bf16 operands:
// packed b128 fragment loads, no per-WMMA cvt. GEMM weight tiles are staged
// into LDS with global_load_async_to_lds_b128 (ASYNCcnt) and shared by all
// 8 waves of a block. Generic GEMM wave tile = 32x64 (8 WMMAs / k-step).
// B=2, S=2048, D=1024, H=16 (hd=64), FF=4096 -> M = 4096.
// ---------------------------------------------------------------------------

typedef __attribute__((ext_vector_type(16))) __bf16 v16bf;
typedef __attribute__((ext_vector_type(8)))  __bf16 v8bf;
typedef __attribute__((ext_vector_type(4)))  __bf16 v4bf;
typedef __attribute__((ext_vector_type(8)))  float  v8f;

__device__ __forceinline__ v16bf cat8(v8bf lo, v8bf hi) {
    return __builtin_shufflevector(lo, hi, 0,1,2,3,4,5,6,7,8,9,10,11,12,13,14,15);
}
// 16 consecutive bf16 (16B-aligned) -> fragment elements 0..15.
__device__ __forceinline__ v16bf loadv16(const __bf16* p) {
    return cat8(*(const v8bf*)p, *(const v8bf*)(p + 8));
}
// A fragment (16x32): lane half hi takes k = hi*16+{0..15} (same contiguous
// k order as the B fragments) -> two b128 loads.
__device__ __forceinline__ v16bf load_a_frag(const __bf16* rowp, int k0, int hi) {
    const __bf16* base = rowp + k0 + hi * 16;
    return cat8(*(const v8bf*)base, *(const v8bf*)(base + 8));
}
__device__ __forceinline__ v8f wmma_bf(v16bf a, v16bf b, v8f c) {
    return __builtin_amdgcn_wmma_f32_16x16x32_bf16(false, a, false, b,
                                                   (short)0, c, false, false);
}
__device__ __forceinline__ void async_b128_to_lds(unsigned ldsaddr, const void* gaddr) {
    asm volatile("global_load_async_to_lds_b128 %0, %1, off"
                 :: "v"(ldsaddr), "v"(gaddr) : "memory");
}

#define LDS_PITCH 40   // bf16 elements per LDS row (80B: 16B aligned, 20-dword bank stride)

// ---------------------------------------------------------------------------
// Generic GEMM: C[M,N] = act(A[M,K]bf16 @ W[N,K]bf16^T + bias) (+ res f32).
// Block = 256 rows x 64 cols (8 waves), wave = 32x64 (2 A-frags, 8 accums).
// W tile (64 cols x 32 k) staged per k-step into LDS via async loads and
// shared by all waves; each LDS B-fragment feeds 2 WMMAs.
// ---------------------------------------------------------------------------
template <bool RELU, bool ADDRES, bool WF32, bool WBF>
__global__ void gemm_bf16_lds(const __bf16* __restrict__ A,
                              const __bf16* __restrict__ W,
                              const float*  __restrict__ bias,
                              const float*  __restrict__ res,
                              float*  __restrict__ Cf,
                              __bf16* __restrict__ Cb,
                              int M, int N, int K) {
    __shared__ __bf16 ldsb[64 * LDS_PITCH];

    const int lane = threadIdx.x & 31;
    const int wave = threadIdx.x >> 5;
    const int hi   = lane >> 4;
    const int ln   = lane & 15;
    const int r0   = blockIdx.y * 256 + wave * 32;
    const int cb0  = blockIdx.x * 64;

    // cooperative B staging: thread t loads 16B (col = t/4, seg = t%4)
    const int tcol = threadIdx.x >> 2;
    const int tseg = threadIdx.x & 3;
    const __bf16* gW = W + (size_t)(cb0 + tcol) * K + tseg * 8;
    const unsigned ldst =
        (unsigned)(uintptr_t)(&ldsb[tcol * LDS_PITCH]) + tseg * 16;

    const __bf16* arow0 = A + (size_t)(r0 + ln) * K;
    const __bf16* arow1 = A + (size_t)(r0 + 16 + ln) * K;

    v8f acc[2][4] = {{{}, {}, {}, {}}, {{}, {}, {}, {}}};
    for (int k0 = 0; k0 < K; k0 += 32) {
        async_b128_to_lds(ldst, gW + k0);
        asm volatile("s_wait_asynccnt 0x0" ::: "memory");
        __syncthreads();

        const v16bf a0 = load_a_frag(arow0, k0, hi);
        const v16bf a1 = load_a_frag(arow1, k0, hi);
#pragma unroll
        for (int t = 0; t < 4; ++t) {
            const __bf16* lb = &ldsb[(t * 16 + ln) * LDS_PITCH + hi * 16];
            const v16bf b = loadv16(lb);
            acc[0][t] = wmma_bf(a0, b, acc[0][t]);
            acc[1][t] = wmma_bf(a1, b, acc[1][t]);
        }
        __syncthreads();           // protect LDS before next overwrite
    }

#pragma unroll
    for (int rr = 0; rr < 2; ++rr) {
#pragma unroll
        for (int t = 0; t < 4; ++t) {
            const int col = cb0 + t * 16 + ln;
            const float bv = bias ? bias[col] : 0.0f;
#pragma unroll
            for (int v = 0; v < 8; ++v) {
                const int row = r0 + rr * 16 + v + (hi << 3);
                float val = acc[rr][t][v] + bv;
                if (RELU)   val = fmaxf(val, 0.0f);
                if (ADDRES) val += res[(size_t)row * N + col];
                if (WF32)   Cf[(size_t)row * N + col] = val;
                if (WBF)    Cb[(size_t)row * N + col] = (__bf16)val;
            }
        }
    }
}

// ---------------------------------------------------------------------------
// QKV GEMM: same pipeline (16-row wave tile), custom epilogue. Writes q/k
// bf16 row-major, k/v caches f32 (reference outputs), and V transposed bf16
// [B, D, S] so the attention P@V B-fragment is a single packed load.
// Batch index is uniform per block (128-row blocks, S % 128 == 0) -> one div.
// ---------------------------------------------------------------------------
__global__ void qkv_gemm(const __bf16* __restrict__ X,
                         const __bf16* __restrict__ Wq,
                         const float*  __restrict__ bqkv,
                         __bf16* __restrict__ qbf,
                         __bf16* __restrict__ kbf,
                         float*  __restrict__ kc,
                         __bf16* __restrict__ vbfT,
                         float*  __restrict__ vc,
                         int M, int D, int S) {
    __shared__ __bf16 ldsb[64 * LDS_PITCH];

    const int lane = threadIdx.x & 31;
    const int wave = threadIdx.x >> 5;
    const int hi   = lane >> 4;
    const int ln   = lane & 15;
    const int K    = D;
    const int r0   = blockIdx.y * 128 + wave * 16;
    const int cb0  = blockIdx.x * 64;

    const int tcol = threadIdx.x >> 2;
    const int tseg = threadIdx.x & 3;
    const __bf16* gW = Wq + (size_t)(cb0 + tcol) * K + tseg * 8;
    const unsigned ldst =
        (unsigned)(uintptr_t)(&ldsb[tcol * LDS_PITCH]) + tseg * 16;

    const __bf16* arow = X + (size_t)(r0 + ln) * K;

    v8f acc[4] = {{}, {}, {}, {}};
    for (int k0 = 0; k0 < K; k0 += 32) {
        async_b128_to_lds(ldst, gW + k0);
        asm volatile("s_wait_asynccnt 0x0" ::: "memory");
        __syncthreads();

        const v16bf a = load_a_frag(arow, k0, hi);
#pragma unroll
        for (int t = 0; t < 4; ++t) {
            const __bf16* lb = &ldsb[(t * 16 + ln) * LDS_PITCH + hi * 16];
            acc[t] = wmma_bf(a, loadv16(lb), acc[t]);
        }
        __syncthreads();
    }

    const int seg  = cb0 / D;       // whole 64-col block lies in q, k, or v
    const int bidx = r0 / S;        // uniform over the block's 128 rows
    const int sr0  = r0 - bidx * S;
#pragma unroll
    for (int t = 0; t < 4; ++t) {
        const int n    = cb0 + t * 16 + ln;
        const int dcol = n - seg * D;
        const float bv = bqkv[n];
#pragma unroll
        for (int v = 0; v < 8; ++v) {
            const int rr  = v + (hi << 3);
            const int row = r0 + rr;
            const float val = acc[t][v] + bv;
            if (seg == 0) {
                qbf[(size_t)row * D + dcol] = (__bf16)val;
            } else if (seg == 1) {
                kc [(size_t)row * D + dcol] = val;
                kbf[(size_t)row * D + dcol] = (__bf16)val;
            } else {
                vc [(size_t)row * D + dcol] = val;
                vbfT[((size_t)bidx * D + dcol) * S + sr0 + rr] = (__bf16)val;
            }
        }
    }
}

// ---------------------------------------------------------------------------
// Causal flash attention, hd = 64, all operands bf16, one q-tile per wave.
// No block barriers (divergent causal trip counts per wave).
// ---------------------------------------------------------------------------
__global__ void attn_kernel(const __bf16* __restrict__ Q,
                            const __bf16* __restrict__ Kb16,
                            const __bf16* __restrict__ VT,
                            __bf16* __restrict__ Ob,
                            int Bv, int S, int Dv, int Hv) {
    const int hd   = 64;
    const int lane = threadIdx.x & 31;
    const int wave = threadIdx.x >> 5;
    const int hi   = lane >> 4;
    const int ln   = lane & 15;

    const int qt_per = S / 16;
    const int g   = blockIdx.x * 8 + wave;
    const int b   = g / (Hv * qt_per);
    const int rem = g % (Hv * qt_per);
    const int h   = rem / qt_per;
    const int qbase = (rem % qt_per) * 16;

    const __bf16* Qb = Q    + ((size_t)b * S) * Dv + h * hd;
    const __bf16* Kk = Kb16 + ((size_t)b * S) * Dv + h * hd;
    const __bf16* Vt = VT   + ((size_t)b * Dv + h * hd) * S;   // [dim][key]

    __shared__ __bf16 plds[8][16][LDS_PITCH];

    float m[8], l[8];
    v8f o[4] = {{}, {}, {}, {}};
#pragma unroll
    for (int v = 0; v < 8; ++v) { m[v] = -1e30f; l[v] = 0.0f; }

    const float scale = 0.125f;          // 1/sqrt(64)
    const __bf16* qrow = Qb + (size_t)(qbase + ln) * Dv;

    for (int kb = 0; kb < qbase + 16; kb += 32) {
        // ---- scores: S[16x32] = Q_tile @ K_chunk^T ----
        v8f s0 = {}, s1 = {};
#pragma unroll
        for (int c = 0; c < 2; ++c) {    // hd = 64 -> two K=32 chunks
            const v16bf aq = load_a_frag(qrow, c * 32, hi);
            const v16bf bk0 =
                loadv16(Kk + (size_t)(kb + ln)      * Dv + c * 32 + hi * 16);
            const v16bf bk1 =
                loadv16(Kk + (size_t)(kb + 16 + ln) * Dv + c * 32 + hi * 16);
            s0 = wmma_bf(aq, bk0, s0);
            s1 = wmma_bf(aq, bk1, s1);
        }

        // ---- scale + causal mask ----
#pragma unroll
        for (int v = 0; v < 8; ++v) {
            const int rowg = qbase + v + (hi << 3);
            float x0 = s0[v] * scale, x1 = s1[v] * scale;
            if (kb + ln      > rowg) x0 = -1e30f;
            if (kb + 16 + ln > rowg) x1 = -1e30f;
            s0[v] = x0; s1[v] = x1;
        }

        // ---- online softmax (row stats via 16-lane xor shuffles) ----
        float mx[8];
#pragma unroll
        for (int v = 0; v < 8; ++v) mx[v] = fmaxf(s0[v], s1[v]);
#pragma unroll
        for (int off = 1; off < 16; off <<= 1)
#pragma unroll
            for (int v = 0; v < 8; ++v)
                mx[v] = fmaxf(mx[v], __shfl_xor(mx[v], off, 32));

        float f[8], rs[8];
#pragma unroll
        for (int v = 0; v < 8; ++v) {
            const float mn = fmaxf(m[v], mx[v]);
            f[v] = __expf(m[v] - mn);
            m[v] = mn;
            const float p0 = __expf(s0[v] - mn);
            const float p1 = __expf(s1[v] - mn);
            s0[v] = p0; s1[v] = p1;
            rs[v] = p0 + p1;
        }
#pragma unroll
        for (int off = 1; off < 16; off <<= 1)
#pragma unroll
            for (int v = 0; v < 8; ++v) rs[v] += __shfl_xor(rs[v], off, 32);
#pragma unroll
        for (int v = 0; v < 8; ++v) {
            l[v] = l[v] * f[v] + rs[v];
            o[0][v] *= f[v]; o[1][v] *= f[v];
            o[2][v] *= f[v]; o[3][v] *= f[v];
        }

        // ---- transpose P (C-layout -> A-layout) through per-wave LDS ----
#pragma unroll
        for (int v = 0; v < 8; ++v) {
            plds[wave][v + (hi << 3)][ln]      = (__bf16)s0[v];
            plds[wave][v + (hi << 3)][ln + 16] = (__bf16)s1[v];
        }
        asm volatile("s_wait_dscnt 0x0" ::: "memory");
        const v16bf pa = load_a_frag(&plds[wave][ln][0], 0, hi);

        // ---- O += P @ V : V^T is [dim][key] -> packed b128 B-fragments ----
#pragma unroll
        for (int t = 0; t < 4; ++t) {
            const v16bf vb =
                loadv16(Vt + (size_t)(t * 16 + ln) * S + kb + hi * 16);
            o[t] = wmma_bf(pa, vb, o[t]);
        }
    }

    // ---- normalize + store bf16 [B,S,D] (feeds out-proj A operand) ----
#pragma unroll
    for (int v = 0; v < 8; ++v) {
        const float inv = (l[v] > 0.0f) ? 1.0f / l[v] : 0.0f;
        const int rowg = qbase + v + (hi << 3);
        __bf16* orow = Ob + ((size_t)b * S + rowg) * Dv + h * hd;
        orow[ln]      = (__bf16)(o[0][v] * inv);
        orow[ln + 16] = (__bf16)(o[1][v] * inv);
        orow[ln + 32] = (__bf16)(o[2][v] * inv);
        orow[ln + 48] = (__bf16)(o[3][v] * inv);
    }
}

// ---------------------------------------------------------------------------
// Row LayerNorm (f32 in, f32 out + optional bf16 copy for next GEMM A).
// ---------------------------------------------------------------------------
__global__ void ln_kernel(const float* __restrict__ X,
                          const float* __restrict__ w,
                          const float* __restrict__ b,
                          float* __restrict__ Y,
                          __bf16* __restrict__ Yb, int D) {
    const int row = blockIdx.x;
    const float* x = X + (size_t)row * D;
    float s = 0.0f, s2 = 0.0f;
    for (int i = threadIdx.x; i < D; i += 256) {
        const float v = x[i]; s += v; s2 += v * v;
    }
#pragma unroll
    for (int off = 16; off > 0; off >>= 1) {
        s  += __shfl_down(s,  off, 32);
        s2 += __shfl_down(s2, off, 32);
    }
    __shared__ float rs[8], rs2[8];
    if ((threadIdx.x & 31) == 0) { rs[threadIdx.x >> 5] = s; rs2[threadIdx.x >> 5] = s2; }
    __syncthreads();
    float ts = 0.0f, ts2 = 0.0f;
#pragma unroll
    for (int i = 0; i < 8; ++i) { ts += rs[i]; ts2 += rs2[i]; }
    const float mu   = ts / (float)D;
    const float var  = ts2 / (float)D - mu * mu;
    const float rstd = rsqrtf(var + 1e-5f);
    float* y = Y + (size_t)row * D;
    for (int i = threadIdx.x; i < D; i += 256) {
        const float val = (x[i] - mu) * rstd * w[i] + b[i];
        y[i] = val;
        if (Yb) Yb[(size_t)row * D + i] = (__bf16)val;
    }
}

// ---------------------------------------------------------------------------
// f32 -> bf16 bulk convert (n multiple of 1024), packed 8B stores.
// ---------------------------------------------------------------------------
__global__ void cvt_bf16(const float* __restrict__ in,
                         __bf16* __restrict__ out, int n) {
    const int i = (blockIdx.x * 256 + threadIdx.x) * 4;
    if (i >= n) return;
    const float4 f = *(const float4*)(in + i);
    v4bf o;
    o[0] = (__bf16)f.x; o[1] = (__bf16)f.y;
    o[2] = (__bf16)f.z; o[3] = (__bf16)f.w;
    *(v4bf*)(out + i) = o;
}

// ---------------------------------------------------------------------------
extern "C" void kernel_launch(void* const* d_in, const int* in_sizes, int n_in,
                              void* d_out, int out_size, void* d_ws, size_t ws_size,
                              hipStream_t stream) {
    const int Bv = 2, Sv = 2048, Dv = 1024, Hv = 16, FFv = 4096;
    const int M = Bv * Sv;                        // 4096
    const size_t MD = (size_t)M * Dv;             // 4M elements

    const float* x      = (const float*)d_in[0];
    const float* qkv_w  = (const float*)d_in[1];
    const float* qkv_b  = (const float*)d_in[2];
    const float* out_w  = (const float*)d_in[3];
    const float* out_b  = (const float*)d_in[4];
    const float* w1     = (const float*)d_in[5];
    const float* b1     = (const float*)d_in[6];
    const float* w2     = (const float*)d_in[7];
    const float* b2     = (const float*)d_in[8];
    const float* ln1w   = (const float*)d_in[9];
    const float* ln1b   = (const float*)d_in[10];
    const float* ln2w   = (const float*)d_in[11];
    const float* ln2b   = (const float*)d_in[12];
    // d_in[13] = attn_mask (causal, analytic), d_in[14] = num_heads

    float* out = (float*)d_out;                   // [M*D]
    float* kc  = out + MD;                        // k_cache f32
    float* vc  = out + 2 * MD;                    // v_cache f32

    // ---- workspace carve-up (bytes) ----
    char* w = (char*)d_ws;
    size_t off = 0;
    auto carve = [&](size_t bytes) { char* p = w + off; off += (bytes + 255) & ~(size_t)255; return p; };
    __bf16* xbf    = (__bf16*)carve(MD * 2);
    __bf16* wqkvbf = (__bf16*)carve((size_t)3 * Dv * Dv * 2);
    __bf16* outwbf = (__bf16*)carve((size_t)Dv * Dv * 2);
    __bf16* w1bf   = (__bf16*)carve((size_t)FFv * Dv * 2);
    __bf16* w2bf   = (__bf16*)carve((size_t)Dv * FFv * 2);
    __bf16* qbf    = (__bf16*)carve(MD * 2);
    __bf16* kbf    = (__bf16*)carve(MD * 2);
    __bf16* vbfT   = (__bf16*)carve(MD * 2);
    __bf16* attnbf = (__bf16*)carve(MD * 2);
    float*  res1   = (float*) carve(MD * 4);
    float*  h1f    = (float*) carve(MD * 4);
    __bf16* h1bf   = (__bf16*)carve(MD * 2);
    __bf16* a1bf   = (__bf16*)carve((size_t)M * FFv * 2);
    float*  res2   = res1;                        // res1 dead after LN1

    const dim3 blk(256);

    // 0) bf16 conversions (x + all weights)
    cvt_bf16<<<dim3(MD / 1024),          blk, 0, stream>>>(x,     xbf,    (int)MD);
    cvt_bf16<<<dim3(3 * Dv * Dv / 1024), blk, 0, stream>>>(qkv_w, wqkvbf, 3 * Dv * Dv);
    cvt_bf16<<<dim3(Dv * Dv / 1024),     blk, 0, stream>>>(out_w, outwbf, Dv * Dv);
    cvt_bf16<<<dim3(FFv * Dv / 1024),    blk, 0, stream>>>(w1,    w1bf,   FFv * Dv);
    cvt_bf16<<<dim3(Dv * FFv / 1024),    blk, 0, stream>>>(w2,    w2bf,   Dv * FFv);

    // 1) QKV projection (writes k_cache / v_cache outputs + bf16 operands)
    qkv_gemm<<<dim3(3 * Dv / 64, M / 128), blk, 0, stream>>>(
        xbf, wqkvbf, qkv_b, qbf, kbf, kc, vbfT, vc, M, Dv, Sv);

    // 2) causal flash attention -> bf16
    attn_kernel<<<dim3((Bv * Hv * (Sv / 16)) / 8), blk, 0, stream>>>(
        qbf, kbf, vbfT, attnbf, Bv, Sv, Dv, Hv);

    // 3) output projection + residual(x) -> res1 f32
    gemm_bf16_lds<false, true, true, false>
        <<<dim3(Dv / 64, M / 256), blk, 0, stream>>>(
        attnbf, outwbf, out_b, x, res1, nullptr, M, Dv, Dv);

    // 4) LayerNorm1 -> h1 (f32 + bf16)
    ln_kernel<<<dim3(M), blk, 0, stream>>>(res1, ln1w, ln1b, h1f, h1bf, Dv);

    // 5) FC1 + ReLU -> a1 bf16 only
    gemm_bf16_lds<true, false, false, true>
        <<<dim3(FFv / 64, M / 256), blk, 0, stream>>>(
        h1bf, w1bf, b1, nullptr, nullptr, a1bf, M, FFv, Dv);

    // 6) FC2 + residual(h1) -> res2 f32
    gemm_bf16_lds<false, true, true, false>
        <<<dim3(Dv / 64, M / 256), blk, 0, stream>>>(
        a1bf, w2bf, b2, h1f, res2, nullptr, M, Dv, FFv);

    // 7) LayerNorm2 -> out
    ln_kernel<<<dim3(M), blk, 0, stream>>>(res2, ln2w, ln2b, out, nullptr, Dv);
}
